// SquadSkim_5179730559405
// MI455X (gfx1250) — compile-verified, matrix operand order checked by reading, not run on hardware
//
#include <hip/hip_runtime.h>
#include <hip/hip_bf16.h>

// ---------------- problem constants ----------------
#define B_    128
#define L_    1024
#define Q_    64
#define E_    256
#define FIX_  16
#define SCAN_ 128
#define CIN_  64    // L/FIX
#define CQ_   4     // Q/FIX
#define EPS_  1e-10f

typedef __bf16 bf16_t;
typedef __attribute__((ext_vector_type(16))) __bf16 v16bf;
typedef __attribute__((ext_vector_type(8)))  __bf16 v8bf;
typedef __attribute__((ext_vector_type(8)))  float  v8f;

static __device__ __forceinline__ bf16_t f2bf(float f) {
  unsigned u = __builtin_bit_cast(unsigned, f);
  unsigned r = (u + 0x7FFFu + ((u >> 16) & 1u)) >> 16;
  return __builtin_bit_cast(bf16_t, (unsigned short)r);
}
static __device__ __forceinline__ float sigm(float x) {
  return 1.0f / (1.0f + __expf(-x));
}

// ---------------- WMMA GEMM: C[M,N] = A[M,K](bf16) x BT[N,K](bf16)^T + bias ----
// One wave computes a 16(M) x 64(N) strip: one A fragment feeds 4 WMMAs
// (register-level A reuse -> 4 wmma per 10 b128 loads). bf16 16x16x32 WMMA,
// f32 accumulate. Fragment layout per CDNA5 ISA (16-bit A 16x32): lane l holds
// row (l&15), K chunks at kb=(l>>4)*8 and kb+16 (two 16B loads per fragment).
union V16U { v16bf v; v8bf h[2]; };

#define WMMA_BF16(A_, B_F, C_) \
  __builtin_amdgcn_wmma_f32_16x16x32_bf16(false, (A_), false, (B_F), (short)0, (C_), false, false)

__global__ void k_wmma_gemm(const bf16_t* __restrict__ A, long lda,
                            const bf16_t* __restrict__ BT,
                            float* __restrict__ C, long ldc,
                            const float* __restrict__ bias,
                            int M, int N, int K) {
  int wid    = threadIdx.x >> 5;
  int lane   = threadIdx.x & 31;
  int nstrip = N >> 6;                       // 64-wide N strips
  int strip  = blockIdx.x * 4 + wid;
  if (strip >= (M >> 4) * nstrip) return;
  int tm = (strip / nstrip) << 4;
  int tn = (strip % nstrip) << 6;
  int row  = lane & 15;
  int half = lane >> 4;
  const long bstr = (long)16 * K;            // stride between 16-col B tiles
  const bf16_t* ap = A  + (long)(tm + row) * lda + (half << 3);
  const bf16_t* bp = BT + (long)(tn + row) * K   + (half << 3);
  v8f acc0 = {}, acc1 = {}, acc2 = {}, acc3 = {};
  for (int k = 0; k < K; k += 32) {
    __builtin_prefetch(ap + k + 128, 0, 1);            // global_prefetch_b8
    __builtin_prefetch(bp + k + 128, 0, 1);
    __builtin_prefetch(bp + 2 * bstr + k + 128, 0, 1);
    V16U a, b0, b1, b2, b3;
    a.h[0]  = *(const v8bf*)(ap + k);
    a.h[1]  = *(const v8bf*)(ap + k + 16);
    b0.h[0] = *(const v8bf*)(bp + k);
    b0.h[1] = *(const v8bf*)(bp + k + 16);
    b1.h[0] = *(const v8bf*)(bp + bstr + k);
    b1.h[1] = *(const v8bf*)(bp + bstr + k + 16);
    b2.h[0] = *(const v8bf*)(bp + 2 * bstr + k);
    b2.h[1] = *(const v8bf*)(bp + 2 * bstr + k + 16);
    b3.h[0] = *(const v8bf*)(bp + 3 * bstr + k);
    b3.h[1] = *(const v8bf*)(bp + 3 * bstr + k + 16);
    acc0 = WMMA_BF16(a.v, b0.v, acc0);
    acc1 = WMMA_BF16(a.v, b1.v, acc1);
    acc2 = WMMA_BF16(a.v, b2.v, acc2);
    acc3 = WMMA_BF16(a.v, b3.v, acc3);
  }
  // store 4 tiles; C layout: lane half picks M rows 0-7 / 8-15, N col = row
  float* crow = C + (long)(tm + (half << 3)) * ldc + tn + row;
  float bv0 = bias ? bias[tn + row]      : 0.0f;
  float bv1 = bias ? bias[tn + 16 + row] : 0.0f;
  float bv2 = bias ? bias[tn + 32 + row] : 0.0f;
  float bv3 = bias ? bias[tn + 48 + row] : 0.0f;
  for (int v = 0; v < 8; ++v) {
    float* cr = crow + (long)v * ldc;
    cr[0]  = acc0[v] + bv0;
    cr[16] = acc1[v] + bv1;
    cr[32] = acc2[v] + bv2;
    cr[48] = acc3[v] + bv3;
  }
}

// ---------------- elementwise / pack kernels ----------------
__global__ void k_zero_f32(float* p, long n) {
  long i = (long)blockIdx.x * blockDim.x + threadIdx.x;
  if (i < n) p[i] = 0.0f;
}
__global__ void k_f2bf_arr(bf16_t* d, const float* s, long n) {
  long i = (long)blockIdx.x * blockDim.x + threadIdx.x;
  if (i < n) d[i] = f2bf(s[i]);
}
__global__ void k_add_bf(bf16_t* d, const float* a, const float* b, long n) {
  long i = (long)blockIdx.x * blockDim.x + threadIdx.x;
  if (i < n) d[i] = f2bf(a[i] + b[i]);
}
__global__ void k_transpose_bf(bf16_t* WT, const float* W, int K, int N) {
  long i = (long)blockIdx.x * blockDim.x + threadIdx.x;
  if (i >= (long)K * N) return;
  int k = (int)(i / N), n = (int)(i % N);
  WT[(long)n * K + k] = f2bf(W[i]);
}

// xh[b,k] = bf16( k<xdim ? emb[ids[b,tok_off + k/E], k%E] : h[b, k-xdim] )
__global__ void k_pack_xh_embed(bf16_t* xh, const float* emb, const int* ids,
                                int ids_stride, int tok_off, int xdim, int Ktot,
                                const float* h) {
  long i = (long)blockIdx.x * blockDim.x + threadIdx.x;
  if (i >= (long)B_ * Ktot) return;
  int b = (int)(i / Ktot), k = (int)(i % Ktot);
  float v;
  if (k < xdim) {
    int tok = k >> 8, e = k & (E_ - 1);
    int id  = ids[b * ids_stride + tok_off + tok];
    v = emb[(long)id * E_ + e];
  } else {
    v = h[b * E_ + (k - xdim)];
  }
  xh[i] = f2bf(v);
}
__global__ void k_pack_xh_f32(bf16_t* xh, const float* x, int xdim, int Ktot,
                              const float* h) {
  long i = (long)blockIdx.x * blockDim.x + threadIdx.x;
  if (i >= (long)B_ * Ktot) return;
  int b = (int)(i / Ktot), k = (int)(i % Ktot);
  float v = (k < xdim) ? x[(long)b * xdim + k] : h[b * E_ + (k - xdim)];
  xh[i] = f2bf(v);
}

// after gate GEMM: r = sigmoid(Gpre[:, :E]); xh[:, xdim:] = bf16(r*h)
__global__ void k_gru_rgate(bf16_t* xh, const float* Gpre, const float* h,
                            int xdim, int Ktot) {
  int i = blockIdx.x * blockDim.x + threadIdx.x;
  if (i >= B_ * E_) return;
  int b = i / E_, e = i % E_;
  float r = sigm(Gpre[b * 2 * E_ + e]);
  xh[(long)b * Ktot + xdim + e] = f2bf(r * h[i]);
}
// u = sigmoid(Gpre[:,E:]); c = tanh(Cpre); h' = u*h+(1-u)*c  (masked)
__global__ void k_gru_update(float* h, float* out, long out_bstride,
                             const float* Gpre, const float* Cpre,
                             const int* seqlen, int div, int t) {
  int i = blockIdx.x * blockDim.x + threadIdx.x;
  if (i >= B_ * E_) return;
  int b = i / E_, e = i % E_;
  bool m = true;
  if (seqlen) m = t < (seqlen[b] / div);
  float u  = sigm(Gpre[b * 2 * E_ + E_ + e]);
  float c  = tanhf(Cpre[i]);
  float ho = h[i];
  float hn = u * ho + (1.0f - u) * c;
  h[i] = m ? hn : ho;
  if (out) out[(long)b * out_bstride + e] = m ? hn : 0.0f;
}

// rep[b,e] = tanh(sum_c outs[b,c,e]*W[c] + b0)
__global__ void k_chunk_dense_tanh(float* rep, const float* outs, const float* W,
                                   const float* b0, int C) {
  int i = blockIdx.x * blockDim.x + threadIdx.x;
  if (i >= B_ * E_) return;
  int b = i / E_, e = i % E_;
  float s = b0[0];
  for (int c = 0; c < C; ++c) s += outs[((long)b * C + c) * E_ + e] * W[c];
  rep[i] = tanhf(s);
}

__global__ void k_lookup(float* look, const float* wrep, const float* qrep,
                         const float* W, const float* bb) {
  int i = blockIdx.x * blockDim.x + threadIdx.x;
  if (i >= B_ * E_) return;
  int b = i / E_, n = i % E_;
  float s = bb[n];
  for (int k = 0; k < E_; ++k) s += wrep[b * E_ + k] * W[k * E_ + n];
  for (int k = 0; k < E_; ++k) s += qrep[b * E_ + k] * W[(E_ + k) * E_ + n];
  look[i] = tanhf(s);
}

__global__ void k_argmax(int* start, float* out0, const float* look) {
  int b = blockIdx.x * blockDim.x + threadIdx.x;
  if (b >= B_) return;
  float best = look[(long)b * E_]; int bi = 0;
  for (int e = 1; e < E_; ++e) {
    float v = look[(long)b * E_ + e];
    if (v > best) { best = v; bi = e; }
  }
  start[b] = bi; out0[b] = (float)bi;
}

__global__ void k_scan_ids(int* scan, const int* input_ids, const int* start) {
  int i = blockIdx.x * blockDim.x + threadIdx.x;
  if (i >= B_ * SCAN_) return;
  int b = i / SCAN_, j = i % SCAN_;
  scan[i] = input_ids[(long)b * L_ + start[b] + j];
}
__global__ void k_rev_ids(int* rev, const int* scan, const int* slen) {
  int i = blockIdx.x * blockDim.x + threadIdx.x;
  if (i >= B_ * SCAN_) return;
  int b = i / SCAN_, t = i % SCAN_;
  int sl = slen[b];
  int idx = (t < sl) ? (sl - 1 - t) : t;
  rev[i] = scan[b * SCAN_ + idx];
}
__global__ void k_reverse_f32(float* out, const float* in, const int* slen) {
  long i = (long)blockIdx.x * blockDim.x + threadIdx.x;
  if (i >= (long)B_ * SCAN_ * E_) return;
  int b = (int)(i / (SCAN_ * E_));
  int r = (int)(i % (SCAN_ * E_));
  int t = r / E_, e = r % E_;
  int sl = slen[b];
  int idx = (t < sl) ? (sl - 1 - t) : t;
  out[i] = in[((long)b * SCAN_ + idx) * E_ + e];
}

// a[b,t] = sum_e tanh(proj[b,t,e] + sproj[b,e]) * W2[e] + b2
__global__ void k_att_score(float* a, const float* proj, const float* sproj,
                            const float* W2, const float* b2, int T) {
  int i = blockIdx.x * blockDim.x + threadIdx.x;
  if (i >= B_ * T) return;
  int b = i / T;
  float s = b2[0];
  const float* pr = proj + (long)i * E_;
  const float* sp = sproj + (long)b * E_;
  for (int e = 0; e < E_; ++e) s += tanhf(pr[e] + sp[e]) * W2[e];
  a[i] = s;
}
// c[b,e] = sum_t ctx[b,t,e]*a[b,t]
__global__ void k_att_ctx(float* c, const float* ctx, const float* a, int T) {
  int i = blockIdx.x * blockDim.x + threadIdx.x;
  if (i >= B_ * E_) return;
  int b = i / E_, e = i % E_;
  float s = 0.0f;
  for (int t = 0; t < T; ++t) s += ctx[((long)b * T + t) * E_ + e] * a[b * T + t];
  c[i] = s;
}
__global__ void k_out_eps(float* out, const float* s, long n) {
  long i = (long)blockIdx.x * blockDim.x + threadIdx.x;
  if (i < n) out[i] = s[i] + EPS_;
}

// ---------------- host side ----------------
static inline dim3 g1(long n, int b = 256) { return dim3((unsigned)((n + b - 1) / b)); }

struct GruW { const bf16_t* WgT; const float* bg; const bf16_t* WcT; const float* bc; };

static void gemm(hipStream_t s, const bf16_t* A, long lda, const bf16_t* BT,
                 float* C, long ldc, const float* bias, int M, int N, int K) {
  int strips = (M >> 4) * (N >> 6);   // 16x64 strip per wave (N % 64 == 0 here)
  k_wmma_gemm<<<dim3((strips + 3) / 4), dim3(128), 0, s>>>(A, lda, BT, C, ldc,
                                                           bias, M, N, K);
}

extern "C" void kernel_launch(void* const* d_in, const int* in_sizes, int n_in,
                              void* d_out, int out_size, void* d_ws, size_t ws_size,
                              hipStream_t stream) {
  (void)in_sizes; (void)n_in; (void)out_size; (void)ws_size;
  #define FIN(i) ((const float*)d_in[(i)])
  #define IIN(i) ((const int*)d_in[(i)])
  // params, insertion order (see reference setup_inputs):
  const float* emb = FIN(0);
  const float* skimI_Wg = FIN(1);  const float* skimI_bg = FIN(2);
  const float* skimI_Wc = FIN(3);  const float* skimI_bc = FIN(4);
  const float* skimI_dW = FIN(5);  const float* skimI_db = FIN(6);
  const float* skimQ_Wg = FIN(7);  const float* skimQ_bg = FIN(8);
  const float* skimQ_Wc = FIN(9);  const float* skimQ_bc = FIN(10);
  const float* skimQ_dW = FIN(11); const float* skimQ_db = FIN(12);
  const float* look_W   = FIN(13); const float* look_b   = FIN(14);
  const float* qf_Wg = FIN(15); const float* qf_bg = FIN(16);
  const float* qf_Wc = FIN(17); const float* qf_bc = FIN(18);
  const float* fw_Wg = FIN(19); const float* fw_bg = FIN(20);
  const float* fw_Wc = FIN(21); const float* fw_bc = FIN(22);
  const float* bw_Wg = FIN(23); const float* bw_bg = FIN(24);
  const float* bw_Wc = FIN(25); const float* bw_bc = FIN(26);
  const float* ad1_W = FIN(27); const float* ad1_b = FIN(28);
  const float* ad2_W = FIN(29); const float* ad2_b = FIN(30);
  const float* ag_Wg = FIN(31); const float* ag_bg = FIN(32);
  const float* ag_Wc = FIN(33); const float* ag_bc = FIN(34);
  const float* pd1_W = FIN(35); const float* pd1_b = FIN(36);
  const float* pd2_W = FIN(37); const float* pd2_b = FIN(38);
  const float* pg_Wg = FIN(39); const float* pg_bg = FIN(40);
  const float* pg_Wc = FIN(41); const float* pg_bc = FIN(42);
  const int* input_ids      = IIN(43);
  const int* question_ids   = IIN(44);
  const int* input_len      = IIN(45);
  const int* input_scan_len = IIN(46);
  const int* question_len   = IIN(47);
  float* out = (float*)d_out;

  // bump allocator over workspace
  char* wsp = (char*)d_ws;
  auto alloc  = [&](size_t bytes) { void* r = wsp; wsp += (bytes + 255) & ~(size_t)255; return r; };
  auto allocF  = [&](size_t n) { return (float*)alloc(n * 4); };
  auto allocBF = [&](size_t n) { return (bf16_t*)alloc(n * 2); };
  auto allocI  = [&](size_t n) { return (int*)alloc(n * 4); };

  auto trans = [&](const float* W, int K, int N) {
    bf16_t* WT = allocBF((size_t)K * N);
    k_transpose_bf<<<g1((long)K * N), 256, 0, stream>>>(WT, W, K, N);
    return WT;
  };

  const int DSK = FIX_ * E_ + E_;   // 4352
  const int D2  = 2 * E_;           // 512

  GruW skimI { trans(skimI_Wg, DSK, D2), skimI_bg, trans(skimI_Wc, DSK, E_), skimI_bc };
  GruW skimQ { trans(skimQ_Wg, DSK, D2), skimQ_bg, trans(skimQ_Wc, DSK, E_), skimQ_bc };
  GruW qfull { trans(qf_Wg, D2, D2), qf_bg, trans(qf_Wc, D2, E_), qf_bc };
  GruW bifw  { trans(fw_Wg, D2, D2), fw_bg, trans(fw_Wc, D2, E_), fw_bc };
  GruW bibw  { trans(bw_Wg, D2, D2), bw_bg, trans(bw_Wc, D2, E_), bw_bc };
  GruW attg  { trans(ag_Wg, D2, D2), ag_bg, trans(ag_Wc, D2, E_), ag_bc };
  GruW predg { trans(pg_Wg, D2, D2), pg_bg, trans(pg_Wc, D2, E_), pg_bc };
  // attn dense d1 split: top rows (ctx part) / bottom rows (state part)
  bf16_t* ad1_topT = trans(ad1_W,            E_, E_);
  bf16_t* ad1_botT = trans(ad1_W + E_ * E_,  E_, E_);
  bf16_t* pd1_topT = trans(pd1_W,            E_, E_);
  bf16_t* pd1_botT = trans(pd1_W + E_ * E_,  E_, E_);

  bf16_t* xh   = allocBF((size_t)B_ * DSK);
  float*  Gpre = allocF((size_t)B_ * D2);
  float*  Cpre = allocF((size_t)B_ * E_);
  float*  h    = allocF((size_t)B_ * E_);

  float* outs_skim = allocF((size_t)B_ * CIN_ * E_);
  float* outs_q    = allocF((size_t)B_ * CQ_ * E_);
  float* word_reps = allocF((size_t)B_ * E_);
  float* q_reps    = allocF((size_t)B_ * E_);
  float* look      = allocF((size_t)B_ * E_);
  int*   start     = allocI(B_);
  int*   scan_ids  = allocI((size_t)B_ * SCAN_);
  int*   rev_ids   = allocI((size_t)B_ * SCAN_);
  float* questions = allocF((size_t)B_ * Q_ * E_);
  bf16_t* questions_bf = allocBF((size_t)B_ * Q_ * E_);
  float* fw      = allocF((size_t)B_ * SCAN_ * E_);
  float* bw_tmp  = allocF((size_t)B_ * SCAN_ * E_);
  float* bw_rev  = allocF((size_t)B_ * SCAN_ * E_);
  bf16_t* inp_bf = allocBF((size_t)B_ * SCAN_ * E_);
  float* Qproj = allocF((size_t)B_ * Q_ * E_);
  float* sproj = allocF((size_t)B_ * E_);
  float* a_buf = allocF((size_t)B_ * Q_);
  float* c_buf = allocF((size_t)B_ * E_);
  float* attention = allocF((size_t)B_ * SCAN_ * E_);
  bf16_t* attention_bf = allocBF((size_t)B_ * SCAN_ * E_);
  float* Aproj = allocF((size_t)B_ * SCAN_ * E_);
  float* st_s  = allocF((size_t)B_ * SCAN_);
  float* ed_s  = allocF((size_t)B_ * SCAN_);
  float* zbuf  = allocF((size_t)B_ * E_);
  bf16_t* s1_bf = allocBF((size_t)B_ * E_);

  const long nBE = (long)B_ * E_;

  // GRU over a token-id sequence; embedding gather fused into the input pack.
  auto run_gru_embed = [&](const GruW& w, const int* ids, int ids_stride,
                           int tok_per_step, const int* seqlen, int div,
                           int T, int xdim, float* outs) {
    int K = xdim + E_;
    k_zero_f32<<<g1(nBE), 256, 0, stream>>>(h, nBE);
    for (int t = 0; t < T; ++t) {
      k_pack_xh_embed<<<g1((long)B_ * K), 256, 0, stream>>>(
          xh, emb, ids, ids_stride, t * tok_per_step, xdim, K, h);
      gemm(stream, xh, K, w.WgT, Gpre, D2, w.bg, B_, D2, K);
      k_gru_rgate<<<g1(nBE), 256, 0, stream>>>(xh, Gpre, h, xdim, K);
      gemm(stream, xh, K, w.WcT, Cpre, E_, w.bc, B_, E_, K);
      k_gru_update<<<g1(nBE), 256, 0, stream>>>(
          h, outs ? outs + (long)t * E_ : nullptr, (long)T * E_,
          Gpre, Cpre, seqlen, div, t);
    }
  };
  // single GRU step with an f32 [B,E] input vector
  auto gru_step_f32 = [&](const GruW& w, const float* x, float* hst,
                          float* outslot, long out_bstride) {
    k_pack_xh_f32<<<g1((long)B_ * D2), 256, 0, stream>>>(xh, x, E_, D2, hst);
    gemm(stream, xh, D2, w.WgT, Gpre, D2, w.bg, B_, D2, D2);
    k_gru_rgate<<<g1(nBE), 256, 0, stream>>>(xh, Gpre, hst, E_, D2);
    gemm(stream, xh, D2, w.WcT, Cpre, E_, w.bc, B_, E_, D2);
    k_gru_update<<<g1(nBE), 256, 0, stream>>>(hst, outslot, out_bstride,
                                              Gpre, Cpre, nullptr, 1, 0);
  };

  // --- skim paragraph / question ---
  run_gru_embed(skimI, input_ids, L_, FIX_, input_len, FIX_, CIN_, FIX_ * E_, outs_skim);
  k_chunk_dense_tanh<<<g1(nBE), 256, 0, stream>>>(word_reps, outs_skim, skimI_dW, skimI_db, CIN_);
  run_gru_embed(skimQ, question_ids, Q_, FIX_, question_len, FIX_, CQ_, FIX_ * E_, outs_q);
  k_chunk_dense_tanh<<<g1(nBE), 256, 0, stream>>>(q_reps, outs_q, skimQ_dW, skimQ_db, CQ_);

  // --- lookup + start offset + scanned span ---
  k_lookup<<<g1(nBE), 256, 0, stream>>>(look, word_reps, q_reps, look_W, look_b);
  k_argmax<<<g1(B_), 128, 0, stream>>>(start, out, look);
  k_scan_ids<<<g1((long)B_ * SCAN_), 256, 0, stream>>>(scan_ids, input_ids, start);
  k_rev_ids<<<g1((long)B_ * SCAN_), 256, 0, stream>>>(rev_ids, scan_ids, input_scan_len);

  // --- full question GRU, bi-GRU over scanned span ---
  run_gru_embed(qfull, question_ids, Q_, 1, question_len, 1, Q_, E_, questions);
  run_gru_embed(bifw, scan_ids, SCAN_, 1, input_scan_len, 1, SCAN_, E_, fw);
  run_gru_embed(bibw, rev_ids, SCAN_, 1, input_scan_len, 1, SCAN_, E_, bw_tmp);
  k_reverse_f32<<<g1((long)B_ * SCAN_ * E_), 256, 0, stream>>>(bw_rev, bw_tmp, input_scan_len);
  k_add_bf<<<g1((long)B_ * SCAN_ * E_), 256, 0, stream>>>(inp_bf, fw, bw_rev, (long)B_ * SCAN_ * E_);

  // --- attention scan (factorized pool: ctx@W_top once, state@W_bot per step) ---
  k_f2bf_arr<<<g1((long)B_ * Q_ * E_), 256, 0, stream>>>(questions_bf, questions, (long)B_ * Q_ * E_);
  gemm(stream, questions_bf, E_, ad1_topT, Qproj, E_, ad1_b, B_ * Q_, E_, E_);
  k_zero_f32<<<g1(nBE), 256, 0, stream>>>(h, nBE);
  for (int t = 0; t < SCAN_; ++t) {
    gemm(stream, inp_bf + (long)t * E_, (long)SCAN_ * E_, ad1_botT, sproj, E_,
         nullptr, B_, E_, E_);
    k_att_score<<<g1((long)B_ * Q_), 256, 0, stream>>>(a_buf, Qproj, sproj, ad2_W, ad2_b, Q_);
    k_att_ctx<<<g1(nBE), 256, 0, stream>>>(c_buf, questions, a_buf, Q_);
    gru_step_f32(attg, c_buf, h, attention + (long)t * E_, (long)SCAN_ * E_);
  }

  // --- answer span: st / ed over the attention sequence ---
  k_f2bf_arr<<<g1((long)B_ * SCAN_ * E_), 256, 0, stream>>>(attention_bf, attention, (long)B_ * SCAN_ * E_);
  gemm(stream, attention_bf, E_, pd1_topT, Aproj, E_, pd1_b, B_ * SCAN_, E_, E_);
  k_zero_f32<<<g1(nBE), 256, 0, stream>>>(zbuf, nBE);      // h0 @ W_bot = 0
  k_att_score<<<g1((long)B_ * SCAN_), 256, 0, stream>>>(st_s, Aproj, zbuf, pd2_W, pd2_b, SCAN_);
  k_att_ctx<<<g1(nBE), 256, 0, stream>>>(c_buf, attention, st_s, SCAN_);
  k_zero_f32<<<g1(nBE), 256, 0, stream>>>(h, nBE);
  gru_step_f32(predg, c_buf, h, nullptr, 0);                // s1 = h
  k_f2bf_arr<<<g1(nBE), 256, 0, stream>>>(s1_bf, h, nBE);
  gemm(stream, s1_bf, E_, pd1_botT, sproj, E_, nullptr, B_, E_, E_);
  k_att_score<<<g1((long)B_ * SCAN_), 256, 0, stream>>>(ed_s, Aproj, sproj, pd2_W, pd2_b, SCAN_);

  // --- outputs: [start(B)] [st+eps(B*SCAN)] [ed+eps(B*SCAN)] ---
  k_out_eps<<<g1((long)B_ * SCAN_), 256, 0, stream>>>(out + B_, st_s, (long)B_ * SCAN_);
  k_out_eps<<<g1((long)B_ * SCAN_), 256, 0, stream>>>(out + B_ + B_ * SCAN_, ed_s, (long)B_ * SCAN_);
  #undef FIN
  #undef IIN
}
